// HierNet_17583596109905
// MI455X (gfx1250) — matrix operand order repaired and checked by reading, not compiled, not run on hardware
//
#include <hip/hip_runtime.h>
#include <hip/hip_bf16.h>

typedef __attribute__((ext_vector_type(16))) _Float16 v16h;
typedef __attribute__((ext_vector_type(8)))  float    v8f;

#define BB   2048
#define NN   32
#define CC   64
#define KK   1000
#define KP   1024   // padded K (64 tiles of 16)
#define DD   1536   // 512 + 8*128
#define NC   2048   // N*C

// ---------------- pack kernels ----------------

__global__ void k_pack_x(const float* __restrict__ x, _Float16* __restrict__ Xh) {
  int i = blockIdx.x * blockDim.x + threadIdx.x;      // B*D = 3145728
  Xh[i] = (_Float16)x[i];
}

__global__ void k_pack_w(const float* __restrict__ W0, const float* __restrict__ W1,
                         _Float16* __restrict__ WhT) {
  int idx = blockIdx.x * blockDim.x + threadIdx.x;    // N*D*C = 3145728, c fastest
  int c = idx & (CC - 1);
  int d = (idx >> 6) % DD;
  int n = idx / (DD * CC);
  float w = (d < 512) ? W0[((n * 512) + d) * CC + c]
                      : W1[((n * 1024) + (d - 512)) * CC + c];
  WhT[((size_t)(n * CC + c)) * DD + d] = (_Float16)w;   // [n][c][d]
}

__global__ void k_pack_bias(const float* __restrict__ b0, const float* __restrict__ b1,
                            float* __restrict__ bias) {
  int i = blockIdx.x * blockDim.x + threadIdx.x;      // N*C = 2048
  int n = i >> 6, c = i & (CC - 1);
  float s = b0[i];
  for (int t = 0; t < 8; ++t) s += b1[(n * 8 + t) * CC + c];
  bias[i] = s;
}

__global__ void k_pack_cw(const float* __restrict__ cw, _Float16* __restrict__ CwT) {
  int idx = blockIdx.x * blockDim.x + threadIdx.x;    // KP*NC = 2097152
  int j = idx & (NC - 1);
  int k = idx >> 11;
  CwT[idx] = (_Float16)((k < KK) ? cw[(size_t)j * KK + k] : 0.0f);  // [k][j]
}

__global__ void k_pack_mask(const float* __restrict__ node_mask, unsigned* __restrict__ bad) {
  int k = blockIdx.x * blockDim.x + threadIdx.x;
  if (k >= KP) return;
  unsigned m = 0;
  if (k < KK) {
    for (int n = 0; n < NN; ++n)
      if (!(node_mask[n * KK + k] >= 1.0f)) m |= (1u << n);  // log(clip)!=0 marker
  }
  bad[k] = m;
}

__global__ void k_gates(const float* __restrict__ gate, const int* __restrict__ anc,
                        float* __restrict__ g, unsigned* __restrict__ zb) {
  int idx = blockIdx.x * blockDim.x + threadIdx.x;    // B*N = 65536
  int b = idx >> 5, n = idx & (NN - 1);
  float p = gate[b * NN + n];
  for (int m = 0; m < NN; ++m)
    if (anc[n * NN + m] != 0) p *= gate[b * NN + m];
  g[idx] = p;                                          // cond_gate * gate
  if (n == 0) {
    unsigned z = 0;
    for (int m = 0; m < NN; ++m)
      if (gate[b * NN + m] == 0.0f) z |= (1u << m);
    zb[b] = z;
  }
}

// ---------------- WMMA helpers ----------------

#define LOAD_AB(FR, PTR, O0, O1)                                   \
  FR.u[0] = *(const uint4*)((PTR) + (O0));                         \
  FR.u[1] = *(const uint4*)((PTR) + (O1));

#define WMMA(ACC, AV, BV)                                          \
  ACC = __builtin_amdgcn_wmma_f32_16x16x32_f16(false, AV, false, BV, (short)0, ACC, false, false);

// ---------------- GEMM1: nout = (X @ Wcat + bias)/5, per node ----------------
// one wave computes a 16x64 tile (4 accumulators sharing one A fragment);
// block = 4 waves covering 64 M rows for one node.

__global__ void k_gemm1(const _Float16* __restrict__ Xh, const _Float16* __restrict__ WhT,
                        const float* __restrict__ bias, float* __restrict__ nout) {
  const int lane = threadIdx.x & 31;
  const int wave = threadIdx.x >> 5;              // 0..3 -> M tile
  const int m0 = blockIdx.x * 64 + wave * 16;
  const int n  = blockIdx.y;
  const int hl = lane >> 4;
  const int r  = lane & 15;

  const _Float16* aRow  = Xh + (size_t)(m0 + r) * DD;
  const _Float16* bBase = WhT + (size_t)(n * CC + r) * DD;   // c-tile ct adds ct*16*DD

  v8f acc0 = {}, acc1 = {}, acc2 = {}, acc3 = {};
  for (int k = 0; k < DD; k += 32) {
    union { v16h v; uint4 u[2]; } A, B0, B1, B2, B3;
    // A: lane holds 16 consecutive K values (lanes 0-15: K 0-15; lanes 16-31: K 16-31)
    LOAD_AB(A, aRow, k + hl * 16, k + hl * 16 + 8)
    // B: lane holds 16 consecutive K values of one column (pre-transposed weights)
    LOAD_AB(B0, bBase + 0 * 16 * DD, k + hl * 16, k + hl * 16 + 8)
    LOAD_AB(B1, bBase + 1 * 16 * DD, k + hl * 16, k + hl * 16 + 8)
    LOAD_AB(B2, bBase + 2 * 16 * DD, k + hl * 16, k + hl * 16 + 8)
    LOAD_AB(B3, bBase + 3 * 16 * DD, k + hl * 16, k + hl * 16 + 8)
    if (k + 32 < DD) {
      __builtin_prefetch(aRow + k + 32, 0, 1);
      __builtin_prefetch(bBase + k + 32, 0, 1);
    }
    WMMA(acc0, A.v, B0.v)
    WMMA(acc1, A.v, B1.v)
    WMMA(acc2, A.v, B2.v)
    WMMA(acc3, A.v, B3.v)
  }
  #pragma unroll
  for (int ct = 0; ct < 4; ++ct) {
    const v8f acc = (ct == 0) ? acc0 : (ct == 1) ? acc1 : (ct == 2) ? acc2 : acc3;
    const int col = ct * 16 + r;
    const float bs = bias[n * CC + col];
    #pragma unroll
    for (int j = 0; j < 8; ++j) {               // C/D layout: VGPR j -> M = j + 8*hl, N = lane&15
      const int row = m0 + j + hl * 8;
      nout[(size_t)row * NC + n * CC + col] = (acc[j] + bs) * 0.2f;
    }
  }
}

// ---------------- scale nout by gates into f16 ----------------

__global__ void k_scale(const float* __restrict__ nout, const float* __restrict__ g,
                        _Float16* __restrict__ SH) {
  int idx = blockIdx.x * blockDim.x + threadIdx.x;    // B*NC = 4194304
  int b = idx >> 11;
  int n = (idx >> 6) & (NN - 1);
  SH[idx] = (_Float16)(nout[idx] * g[b * NN + n]);
}

// ---------------- GEMM2: output = SH(2048x2048) @ CW(2048x1024pad) ----------------
// one wave computes a 16x64 output tile (4 k-tiles sharing one A fragment).

__global__ void k_gemm2(const _Float16* __restrict__ SH, const _Float16* __restrict__ CwT,
                        float* __restrict__ out) {
  const int lane = threadIdx.x & 31;
  const int wave = threadIdx.x >> 5;              // 0..3
  const int grp  = blockIdx.y * 4 + wave;         // 0..15, covers 64 k columns
  const int kt0  = grp * 64;
  const int m0 = blockIdx.x * 16;
  const int hl = lane >> 4;
  const int r  = lane & 15;

  const _Float16* aRow  = SH + (size_t)(m0 + r) * NC;
  const _Float16* bBase = CwT + (size_t)(kt0 + r) * NC;      // q adds q*16*NC

  v8f acc0 = {}, acc1 = {}, acc2 = {}, acc3 = {};
  for (int j = 0; j < NC; j += 32) {
    union { v16h v; uint4 u[2]; } A, B0, B1, B2, B3;
    LOAD_AB(A, aRow, j + hl * 16, j + hl * 16 + 8)
    LOAD_AB(B0, bBase + 0 * 16 * NC, j + hl * 16, j + hl * 16 + 8)
    LOAD_AB(B1, bBase + 1 * 16 * NC, j + hl * 16, j + hl * 16 + 8)
    LOAD_AB(B2, bBase + 2 * 16 * NC, j + hl * 16, j + hl * 16 + 8)
    LOAD_AB(B3, bBase + 3 * 16 * NC, j + hl * 16, j + hl * 16 + 8)
    if (j + 32 < NC) {
      __builtin_prefetch(aRow + j + 32, 0, 1);
      __builtin_prefetch(bBase + j + 32, 0, 1);
    }
    WMMA(acc0, A.v, B0.v)
    WMMA(acc1, A.v, B1.v)
    WMMA(acc2, A.v, B2.v)
    WMMA(acc3, A.v, B3.v)
  }
  #pragma unroll
  for (int q = 0; q < 4; ++q) {
    const v8f acc = (q == 0) ? acc0 : (q == 1) ? acc1 : (q == 2) ? acc2 : acc3;
    const int kk = kt0 + q * 16 + r;
    if (kk < KK) {
      #pragma unroll
      for (int j = 0; j < 8; ++j)
        out[(size_t)(m0 + j + hl * 8) * KK + kk] = acc[j];
    }
  }
}

// ---------------- masked softmax base ----------------

__global__ void k_sfmx(const float* __restrict__ out, const unsigned* __restrict__ zb,
                       const unsigned* __restrict__ bad, float* __restrict__ sb) {
  __shared__ float red[256];
  const int b = blockIdx.x;
  const unsigned z = zb[b];
  float s = 0.0f;
  for (int k = threadIdx.x; k < KK; k += 256)
    if ((z & bad[k]) == 0u) s += expf(out[(size_t)b * KK + k]);
  red[threadIdx.x] = s;
  __syncthreads();
  for (int off = 128; off > 0; off >>= 1) {
    if (threadIdx.x < off) red[threadIdx.x] += red[threadIdx.x + off];
    __syncthreads();
  }
  if (threadIdx.x == 0) sb[b] = red[0];
}

// ---------------- launch ----------------

extern "C" void kernel_launch(void* const* d_in, const int* in_sizes, int n_in,
                              void* d_out, int out_size, void* d_ws, size_t ws_size,
                              hipStream_t stream) {
  const float* x    = (const float*)d_in[0];
  const float* gate = (const float*)d_in[1];
  const float* W0   = (const float*)d_in[2];
  const float* b0   = (const float*)d_in[3];
  const float* W1   = (const float*)d_in[4];
  const float* b1   = (const float*)d_in[5];
  const float* cw   = (const float*)d_in[6];
  const float* nm   = (const float*)d_in[7];
  const int*   anc  = (const int*)d_in[8];

  float* out       = (float*)d_out;                   // (B,K)
  float* nout      = out + (size_t)BB * KK;           // (B,N,C)
  float* sfmx_base = nout + (size_t)BB * NC;          // (B,)

  char* ws = (char*)d_ws;
  _Float16* Xh   = (_Float16*)(ws);                                   // 6291456 B
  _Float16* WhT  = (_Float16*)(ws + 6291456);                         // 6291456 B
  _Float16* CwT  = (_Float16*)(ws + 12582912);                        // 4194304 B
  _Float16* SH   = (_Float16*)(ws + 16777216);                        // 8388608 B
  float*    bias = (float*)(ws + 25165824);                           // 8192 B
  float*    g    = (float*)(ws + 25174016);                           // 262144 B
  unsigned* zb   = (unsigned*)(ws + 25436160);                        // 8192 B
  unsigned* bad  = (unsigned*)(ws + 25444352);                        // 4096 B

  k_pack_x   <<<(BB * DD) / 256, 256, 0, stream>>>(x, Xh);
  k_pack_w   <<<(NN * DD * CC) / 256, 256, 0, stream>>>(W0, W1, WhT);
  k_pack_bias<<<(NN * CC) / 256, 256, 0, stream>>>(b0, b1, bias);
  k_pack_cw  <<<(KP * NC) / 256, 256, 0, stream>>>(cw, CwT);
  k_pack_mask<<<(KP + 255) / 256, 256, 0, stream>>>(nm, bad);
  k_gates    <<<(BB * NN) / 256, 256, 0, stream>>>(gate, anc, g, zb);

  k_gemm1<<<dim3(BB / 64, NN), 128, 0, stream>>>(Xh, WhT, bias, nout);
  k_scale<<<(BB * NC) / 256, 256, 0, stream>>>(nout, g, SH);
  k_gemm2<<<dim3(BB / 16, 4), 128, 0, stream>>>(SH, CwT, out);
  k_sfmx <<<BB, 256, 0, stream>>>(out, zb, bad, sfmx_base);
}
